// Decoder_19112604467862
// MI455X (gfx1250) — matrix operand order repaired and checked by reading, not compiled, not run on hardware
//
#include <hip/hip_runtime.h>
#include <hip/hip_bf16.h>
#include <math.h>

#define BSZ 8
#define TGT 64
#define SRC 512
#define AGENTS 4
#define HSZ 512
#define EMB 256
#define VOCAB 32000

typedef __attribute__((ext_vector_type(16))) __bf16 bf16x16;
typedef __attribute__((ext_vector_type(4)))  __bf16 bf16x4;
typedef __attribute__((ext_vector_type(8)))  float  floatx8;

__device__ __forceinline__ float sigmoidf_(float x) { return 1.0f / (1.0f + __expf(-x)); }

// ---- WMMA fragment helpers (layouts per CDNA5 ISA 7.12.2, wave32) ----

// A-matrix 16x32 bf16: lane = row M (lane&15); lanes<16 hold K{0..7,16..23},
// lanes>=16 hold K{8..15,24..31}. base points at (row0, k0); ld in elements.
__device__ __forceinline__ bf16x16 load_a_frag(const __bf16* base, int ld) {
  const int lane = threadIdx.x & 31;
  const int m = lane & 15;
  const int kb = (lane < 16) ? 0 : 8;
  const __bf16* p = base + m * ld + kb;
  bf16x16 a;
#pragma unroll
  for (int i = 0; i < 8; ++i) a[i] = p[i];
#pragma unroll
  for (int i = 0; i < 8; ++i) a[8 + i] = p[16 + i];
  return a;
}

// B-matrix 32x16 bf16: N = lane&15; lanes<16 hold K{0..7,16..23}, lanes>=16 K{8..15,24..31}.
// "nmajor": memory holds B^T (i.e. W[n][k] row-major); element (K,N) at base[N*ld+K].
__device__ __forceinline__ bf16x16 load_b_frag_nmajor(const __bf16* base, int ld) {
  const int lane = threadIdx.x & 31;
  const int n = lane & 15;
  const int kb = (lane < 16) ? 0 : 8;
  const __bf16* p = base + n * ld + kb;
  bf16x16 b;
#pragma unroll
  for (int i = 0; i < 8; ++i) b[i] = p[i];
#pragma unroll
  for (int i = 0; i < 8; ++i) b[8 + i] = p[16 + i];
  return b;
}

__device__ __forceinline__ floatx8 wmma_bf16(bf16x16 a, bf16x16 b, floatx8 c) {
  return __builtin_amdgcn_wmma_f32_16x16x32_bf16(false, a, false, b, (short)0, c,
                                                 false, false);
}

// ---- conversion kernels (vectorized: b128 in, b64 out) ----
__global__ void cvt_bf16x4_kernel(const float4* __restrict__ src,
                                  bf16x4* __restrict__ dst, int n4) {
  int i = blockIdx.x * 256 + threadIdx.x;
  if (i < n4) {
    float4 v = src[i];
    bf16x4 o = {(__bf16)v.x, (__bf16)v.y, (__bf16)v.z, (__bf16)v.w};
    dst[i] = o;
  }
}

// encoded_seq [SRC,BSZ,AGENTS,HSZ] f32 -> enc_bf [BSZ,AGENTS,SRC,HSZ] bf16
__global__ void enc_cvt_kernel(const float4* __restrict__ src, bf16x4* __restrict__ dst) {
  int idx = blockIdx.x * 256 + threadIdx.x;  // 2^21 float4 groups exactly
  int h4 = idx & 127;
  int s = (idx >> 7) & 511;
  int a = (idx >> 16) & 3;
  int b = idx >> 18;
  float4 v = src[(((s * BSZ + b) * AGENTS + a) * HSZ >> 2) + h4];
  bf16x4 o = {(__bf16)v.x, (__bf16)v.y, (__bf16)v.z, (__bf16)v.w};
  dst[idx] = o;
}

// enc_bf [b,a,s,h] -> enc_t [b,a,h,s]  (LDS tiled transpose, both sides coalesced)
__global__ void enc_transpose_kernel(const __bf16* __restrict__ enc_bf,
                                     __bf16* __restrict__ enc_t) {
  __shared__ __bf16 tile[32][34];
  const int pair = blockIdx.x >> 8;  // b*AGENTS+a
  const int rem = blockIdx.x & 255;  // 16x16 grid of 32x32 tiles
  const int s0 = (rem & 15) * 32;
  const int h0 = (rem >> 4) * 32;
  const int tx = threadIdx.x & 31;
  const int ty = threadIdx.x >> 5;  // 8 rows, 4 iterations
  const __bf16* src = enc_bf + (size_t)pair * SRC * HSZ;
  __bf16* dst = enc_t + (size_t)pair * HSZ * SRC;
#pragma unroll
  for (int i = 0; i < 4; ++i)
    tile[ty + i * 8][tx] = src[(size_t)(s0 + ty + i * 8) * HSZ + h0 + tx];
  __syncthreads();
#pragma unroll
  for (int i = 0; i < 4; ++i)
    dst[(size_t)(h0 + ty + i * 8) * SRC + s0 + tx] = tile[tx][ty + i * 8];
}

// ---- LSTM: single persistent workgroup (sequence is serial) ----
__global__ void __launch_bounds__(512)
lstm_kernel(const float* __restrict__ prev_input, const float* __restrict__ init_h,
            const float* __restrict__ init_c, const __bf16* __restrict__ Wih,
            const __bf16* __restrict__ Whh, const float* __restrict__ b_lstm,
            float* __restrict__ state_f, __bf16* __restrict__ state_bf) {
  __shared__ __bf16 xs[16 * EMB];        // 8 KB  (rows 8..15 zero)
  __shared__ __bf16 hs[16 * HSZ];        // 16 KB (rows 8..15 zero)
  __shared__ float  gates[8 * 4 * HSZ];  // 64 KB
  __shared__ float  cs[8 * HSZ];         // 16 KB

  const int tid = threadIdx.x;
  const int lane = tid & 31;
  const int wave = tid >> 5;

  for (int i = tid; i < 16 * HSZ; i += 512) {
    int r = i / HSZ, col = i % HSZ;
    hs[i] = (r < BSZ) ? (__bf16)init_h[r * HSZ + col] : (__bf16)0.0f;
  }
  for (int i = tid; i < 8 * HSZ; i += 512) cs[i] = init_c[i];
  for (int i = tid; i < 16 * EMB; i += 512)
    if (i >= 8 * EMB) xs[i] = (__bf16)0.0f;
  __syncthreads();

  for (int t = 0; t < TGT; ++t) {
    for (int i = tid; i < 8 * EMB; i += 512) {
      int r = i / EMB, col = i % EMB;
      xs[r * EMB + col] = (__bf16)prev_input[(r * TGT + t) * EMB + col];
    }
    __syncthreads();

    // gates[0:8, 0:2048] = x @ Wih^T + h @ Whh^T  (128 N-tiles over 16 waves)
    for (int j = 0; j < 8; ++j) {
      const int nt = wave * 8 + j;
      floatx8 acc = {};
      for (int kf = 0; kf < EMB / 32; ++kf) {
        bf16x16 a = load_a_frag(xs + kf * 32, EMB);
        bf16x16 b = load_b_frag_nmajor(Wih + (nt * 16) * EMB + kf * 32, EMB);
        acc = wmma_bf16(a, b, acc);
      }
      for (int kf = 0; kf < HSZ / 32; ++kf) {
        bf16x16 a = load_a_frag(hs + kf * 32, HSZ);
        bf16x16 b = load_b_frag_nmajor(Whh + (nt * 16) * HSZ + kf * 32, HSZ);
        acc = wmma_bf16(a, b, acc);
      }
      if (lane < 16) {  // lanes 0..15 hold rows M=0..7 (valid batch rows)
#pragma unroll
        for (int r = 0; r < 8; ++r) gates[r * (4 * HSZ) + nt * 16 + lane] = acc[r];
      }
    }
    __syncthreads();

    for (int i = tid; i < 8 * HSZ; i += 512) {
      int r = i / HSZ, col = i % HSZ;
      float ig = gates[r * 4 * HSZ + col] + b_lstm[col];
      float fg = gates[r * 4 * HSZ + HSZ + col] + b_lstm[HSZ + col];
      float gg = gates[r * 4 * HSZ + 2 * HSZ + col] + b_lstm[2 * HSZ + col];
      float og = gates[r * 4 * HSZ + 3 * HSZ + col] + b_lstm[3 * HSZ + col];
      float c = sigmoidf_(fg) * cs[i] + sigmoidf_(ig) * tanhf(gg);
      float h = sigmoidf_(og) * tanhf(c);
      cs[i] = c;
      hs[r * HSZ + col] = (__bf16)h;
      state_f[(r * TGT + t) * HSZ + col] = h;
      state_bf[(r * TGT + t) * HSZ + col] = (__bf16)h;
    }
    __syncthreads();
  }
}

// ---- word attention: one block per (b, agent) ----
__global__ void __launch_bounds__(256)
word_attn_kernel(const __bf16* __restrict__ state_bf, const __bf16* __restrict__ enc_bf,
                 const __bf16* __restrict__ enc_t, float* __restrict__ attn_f,
                 __bf16* __restrict__ attn_bf, float* __restrict__ ctx_f) {
  __shared__ float scores[TGT * SRC];  // 128 KB
  const int blk = blockIdx.x;          // b*AGENTS + a
  const int bi = blk >> 2, ai = blk & 3;
  const int tid = threadIdx.x;
  const int wave = tid >> 5;
  const int lane = tid & 31;
  const float scale = 0.044194173824159216f;  // 1/sqrt(512)

  // scores[t][s] = state[t]·enc[s] * scale
  for (int tile = wave; tile < 128; tile += 8) {
    const int mt = tile >> 5, st = tile & 31;
    floatx8 acc = {};
    for (int kf = 0; kf < HSZ / 32; ++kf) {
      bf16x16 a = load_a_frag(state_bf + (bi * TGT + mt * 16) * HSZ + kf * 32, HSZ);
      bf16x16 b =
          load_b_frag_nmajor(enc_bf + ((size_t)blk * SRC + st * 16) * HSZ + kf * 32, HSZ);
      acc = wmma_bf16(a, b, acc);
    }
    const int n = lane & 15, mb = (lane < 16) ? 0 : 8;
#pragma unroll
    for (int r = 0; r < 8; ++r)
      scores[(mt * 16 + mb + r) * SRC + st * 16 + n] = acc[r] * scale;
  }
  __syncthreads();

  // softmax over s per row
  if (tid < TGT) {
    const int row = tid;
    float m = -1e30f;
    for (int s = 0; s < SRC; ++s) m = fmaxf(m, scores[row * SRC + s]);
    float sum = 0.f;
    for (int s = 0; s < SRC; ++s) sum += __expf(scores[row * SRC + s] - m);
    const float inv = 1.0f / sum;
    for (int s = 0; s < SRC; ++s) {
      float p = __expf(scores[row * SRC + s] - m) * inv;
      attn_f[((size_t)blk * TGT + row) * SRC + s] = p;
      attn_bf[((size_t)blk * TGT + row) * SRC + s] = (__bf16)p;
    }
  }
  __threadfence_block();
  __syncthreads();

  // ctx[t][h] = attn[t]·enc[:,h]  (B from transposed enc_t -> contiguous frag loads)
  for (int tile = wave; tile < 128; tile += 8) {
    const int mt = tile >> 5, ht = tile & 31;
    floatx8 acc = {};
    for (int kf = 0; kf < SRC / 32; ++kf) {
      bf16x16 a = load_a_frag(attn_bf + ((size_t)blk * TGT + mt * 16) * SRC + kf * 32, SRC);
      bf16x16 b =
          load_b_frag_nmajor(enc_t + ((size_t)blk * HSZ + ht * 16) * SRC + kf * 32, SRC);
      acc = wmma_bf16(a, b, acc);
    }
    const int n = lane & 15, mb = (lane < 16) ? 0 : 8;
#pragma unroll
    for (int r = 0; r < 8; ++r) {
      const int tIdx = mt * 16 + mb + r;
      ctx_f[((bi * TGT + tIdx) * AGENTS + ai) * HSZ + ht * 16 + n] = acc[r];
    }
  }
}

// ---- agent attention + pointer-gen gate + concat row build: one block per (b,t) ----
__global__ void __launch_bounds__(128)
agent_gen_kernel(const float* __restrict__ state_f, const float* __restrict__ ctx_f,
                 const float* __restrict__ prev_input, const float* __restrict__ context_w,
                 const float* __restrict__ sp_w, const float* __restrict__ sp_b,
                 float* __restrict__ gen_ws, __bf16* __restrict__ cat_bf,
                 float* __restrict__ out_agent_attn) {
  __shared__ float red[128];
  const int row = blockIdx.x;  // b*TGT + t
  const int tid = threadIdx.x;
  const float* st = state_f + (size_t)row * HSZ;
  const float* cx = ctx_f + (size_t)row * AGENTS * HSZ;

  auto block_reduce = [&](float v) -> float {
    red[tid] = v;
    __syncthreads();
    for (int off = 64; off > 0; off >>= 1) {
      if (tid < off) red[tid] += red[tid + off];
      __syncthreads();
    }
    float r = red[0];
    __syncthreads();
    return r;
  };

  const float scale = 0.044194173824159216f;
  float attn[AGENTS], gen[AGENTS];
  for (int a = 0; a < AGENTS; ++a) {
    float p = 0.f;
    for (int h = tid; h < HSZ; h += 128) p += st[h] * cx[a * HSZ + h];
    attn[a] = block_reduce(p) * scale;
    float q = 0.f;
    for (int h = tid; h < HSZ; h += 128) q += cx[a * HSZ + h] * context_w[h];
    gen[a] = block_reduce(q);  // ctx importance
  }
  float sp = 0.f;
  for (int i = tid; i < HSZ + EMB; i += 128)
    sp += (i < HSZ ? st[i] : prev_input[(size_t)row * EMB + (i - HSZ)]) * sp_w[i];
  sp = block_reduce(sp) + sp_b[0];

  float m = fmaxf(fmaxf(attn[0], attn[1]), fmaxf(attn[2], attn[3]));
  float ssum = 0.f;
  for (int a = 0; a < AGENTS; ++a) { attn[a] = __expf(attn[a] - m); ssum += attn[a]; }
  const float sinv = 1.0f / ssum;
  for (int a = 0; a < AGENTS; ++a) {
    attn[a] *= sinv;
    gen[a] = sigmoidf_(gen[a] + sp);
  }

  if (tid < AGENTS) {
    out_agent_attn[row * AGENTS + tid] = attn[tid];
    gen_ws[row * AGENTS + tid] = gen[tid];
  }
  for (int h = tid; h < HSZ; h += 128) {
    float g = attn[0] * cx[h] + attn[1] * cx[HSZ + h] + attn[2] * cx[2 * HSZ + h] +
              attn[3] * cx[3 * HSZ + h];
    cat_bf[(size_t)row * (2 * HSZ) + h] = (__bf16)st[h];
    cat_bf[(size_t)row * (2 * HSZ) + HSZ + h] = (__bf16)g;
  }
}

// ---- vocab GEMM: each wave owns 1 N-tile x 8 M-tiles (8 accumulators) so every
// B-fragment (the 65MB W_vocab stream) is reused 8x; A (1MB) stays cache-resident.
__global__ void __launch_bounds__(256)
vocab_gemm_kernel(const __bf16* __restrict__ cat_bf, const __bf16* __restrict__ wv_bf,
                  const float* __restrict__ b_vocab, float* __restrict__ logits) {
  const int wave = threadIdx.x >> 5;
  const int lane = threadIdx.x & 31;
  const int task = blockIdx.x * 8 + wave;  // 8000 tasks
  const int nt = task >> 2;                // 0..1999
  const int mg = task & 3;                 // m-group: rows mg*128 .. mg*128+127

  floatx8 acc[8];
#pragma unroll
  for (int g = 0; g < 8; ++g) acc[g] = (floatx8){};

  for (int kf = 0; kf < (2 * HSZ) / 32; ++kf) {
    bf16x16 b = load_b_frag_nmajor(wv_bf + (size_t)(nt * 16) * (2 * HSZ) + kf * 32, 2 * HSZ);
#pragma unroll
    for (int g = 0; g < 8; ++g) {
      bf16x16 a =
          load_a_frag(cat_bf + (size_t)((mg * 8 + g) * 16) * (2 * HSZ) + kf * 32, 2 * HSZ);
      acc[g] = wmma_bf16(a, b, acc[g]);
    }
  }
  const int n = lane & 15, mb = (lane < 16) ? 0 : 8;
  const float bias = b_vocab[nt * 16 + n];
#pragma unroll
  for (int g = 0; g < 8; ++g) {
#pragma unroll
    for (int r = 0; r < 8; ++r)
      logits[(size_t)((mg * 8 + g) * 16 + mb + r) * VOCAB + nt * 16 + n] = acc[g][r] + bias;
  }
}

// ---- fused row softmax + gen-scaled output expansion (262 MB stream, float4) ----
__global__ void __launch_bounds__(256)
vocab_out_kernel(const float* __restrict__ logits, const float* __restrict__ gen_ws,
                 float* __restrict__ out_voc) {
  __shared__ float red[256];
  const int row = blockIdx.x;  // 512 rows (b*TGT+t)
  const int tid = threadIdx.x;
  const float4* l4 = (const float4*)(logits + (size_t)row * VOCAB);
  const int N4 = VOCAB / 4;  // 8000

  float m = -1e30f;
  for (int v = tid; v < N4; v += 256) {
    float4 x = l4[v];
    m = fmaxf(m, fmaxf(fmaxf(x.x, x.y), fmaxf(x.z, x.w)));
  }
  red[tid] = m;
  __syncthreads();
  for (int off = 128; off > 0; off >>= 1) {
    if (tid < off) red[tid] = fmaxf(red[tid], red[tid + off]);
    __syncthreads();
  }
  m = red[0];
  __syncthreads();
  float s = 0.f;
  for (int v = tid; v < N4; v += 256) {
    float4 x = l4[v];
    s += __expf(x.x - m) + __expf(x.y - m) + __expf(x.z - m) + __expf(x.w - m);
  }
  red[tid] = s;
  __syncthreads();
  for (int off = 128; off > 0; off >>= 1) {
    if (tid < off) red[tid] += red[tid + off];
    __syncthreads();
  }
  const float inv = 1.0f / red[0];
  const float g0 = gen_ws[row * 4 + 0], g1 = gen_ws[row * 4 + 1];
  const float g2 = gen_ws[row * 4 + 2], g3 = gen_ws[row * 4 + 3];
  float4* o4 = (float4*)(out_voc + (size_t)row * AGENTS * VOCAB);
  for (int v = tid; v < N4; v += 256) {
    float4 x = l4[v];
    float4 p = {__expf(x.x - m) * inv, __expf(x.y - m) * inv, __expf(x.z - m) * inv,
                __expf(x.w - m) * inv};
    o4[v] = (float4){g0 * p.x, g0 * p.y, g0 * p.z, g0 * p.w};
    o4[N4 + v] = (float4){g1 * p.x, g1 * p.y, g1 * p.z, g1 * p.w};
    o4[2 * N4 + v] = (float4){g2 * p.x, g2 * p.y, g2 * p.z, g2 * p.w};
    o4[3 * N4 + v] = (float4){g3 * p.x, g3 * p.y, g3 * p.z, g3 * p.w};
  }
}

// ---- copy_prob[b][t][a][s] = (1-gen) * attn  (float4) ----
__global__ void copy_prob_kernel(const float4* __restrict__ attn_f4,
                                 const float* __restrict__ gen_ws,
                                 float4* __restrict__ out_copy4) {
  int idx = blockIdx.x * 256 + threadIdx.x;  // 262144 float4 groups exactly
  int s4 = idx & 127;
  int a = (idx >> 7) & 3;
  int t = (idx >> 9) & 63;
  int b = idx >> 15;
  int row = b * TGT + t;
  float g = 1.0f - gen_ws[row * AGENTS + a];
  float4 x = attn_f4[(((size_t)(b * AGENTS + a) * TGT + t) * SRC >> 2) + s4];
  out_copy4[idx] = (float4){g * x.x, g * x.y, g * x.z, g * x.w};
}

extern "C" void kernel_launch(void* const* d_in, const int* in_sizes, int n_in,
                              void* d_out, int out_size, void* d_ws, size_t ws_size,
                              hipStream_t stream) {
  const float* prev_input = (const float*)d_in[0];
  const float* encoded    = (const float*)d_in[1];
  const float* init_h     = (const float*)d_in[2];
  const float* init_c     = (const float*)d_in[3];
  const float* W_ih       = (const float*)d_in[4];
  const float* W_hh       = (const float*)d_in[5];
  const float* b_lstm     = (const float*)d_in[6];
  const float* W_vocab    = (const float*)d_in[7];
  const float* b_vocab    = (const float*)d_in[8];
  const float* context_w  = (const float*)d_in[9];
  const float* sp_w       = (const float*)d_in[10];
  const float* sp_b       = (const float*)d_in[11];

  char* ws = (char*)d_ws;
  size_t off = 0;
  auto walloc = [&](size_t bytes) -> void* {
    void* p = ws + off;
    off += (bytes + 255) & ~(size_t)255;
    return p;
  };
  __bf16* enc_bf   = (__bf16*)walloc((size_t)BSZ * AGENTS * SRC * HSZ * 2);
  __bf16* enc_t    = (__bf16*)walloc((size_t)BSZ * AGENTS * HSZ * SRC * 2);
  __bf16* wih_bf   = (__bf16*)walloc((size_t)4 * HSZ * EMB * 2);
  __bf16* whh_bf   = (__bf16*)walloc((size_t)4 * HSZ * HSZ * 2);
  __bf16* wv_bf    = (__bf16*)walloc((size_t)VOCAB * 2 * HSZ * 2);
  float*  state_f  = (float*)walloc((size_t)BSZ * TGT * HSZ * 4);
  __bf16* state_bf = (__bf16*)walloc((size_t)BSZ * TGT * HSZ * 2);
  float*  attn_f   = (float*)walloc((size_t)BSZ * AGENTS * TGT * SRC * 4);
  __bf16* attn_bf  = (__bf16*)walloc((size_t)BSZ * AGENTS * TGT * SRC * 2);
  float*  ctx_f    = (float*)walloc((size_t)BSZ * TGT * AGENTS * HSZ * 4);
  float*  gen_ws   = (float*)walloc((size_t)BSZ * TGT * AGENTS * 4);
  __bf16* cat_bf   = (__bf16*)walloc((size_t)BSZ * TGT * 2 * HSZ * 2);
  float*  logits   = (float*)walloc((size_t)BSZ * TGT * VOCAB * 4);

  float* out_voc   = (float*)d_out;
  float* out_copy  = out_voc + (size_t)BSZ * TGT * AGENTS * VOCAB;
  float* out_agent = out_copy + (size_t)BSZ * TGT * AGENTS * SRC;

  cvt_bf16x4_kernel<<<(4 * HSZ * EMB / 4 + 255) / 256, 256, 0, stream>>>(
      (const float4*)W_ih, (bf16x4*)wih_bf, 4 * HSZ * EMB / 4);
  cvt_bf16x4_kernel<<<(4 * HSZ * HSZ / 4 + 255) / 256, 256, 0, stream>>>(
      (const float4*)W_hh, (bf16x4*)whh_bf, 4 * HSZ * HSZ / 4);
  cvt_bf16x4_kernel<<<(VOCAB * 2 * HSZ / 4 + 255) / 256, 256, 0, stream>>>(
      (const float4*)W_vocab, (bf16x4*)wv_bf, VOCAB * 2 * HSZ / 4);
  enc_cvt_kernel<<<(BSZ * AGENTS * SRC * HSZ / 4) / 256, 256, 0, stream>>>(
      (const float4*)encoded, (bf16x4*)enc_bf);
  enc_transpose_kernel<<<BSZ * AGENTS * 256, 256, 0, stream>>>(enc_bf, enc_t);

  lstm_kernel<<<1, 512, 0, stream>>>(prev_input, init_h, init_c, wih_bf, whh_bf,
                                     b_lstm, state_f, state_bf);
  word_attn_kernel<<<BSZ * AGENTS, 256, 0, stream>>>(state_bf, enc_bf, enc_t, attn_f,
                                                     attn_bf, ctx_f);
  agent_gen_kernel<<<BSZ * TGT, 128, 0, stream>>>(state_f, ctx_f, prev_input, context_w,
                                                  sp_w, sp_b, gen_ws, cat_bf, out_agent);
  vocab_gemm_kernel<<<1000, 256, 0, stream>>>(cat_bf, wv_bf, b_vocab, logits);
  vocab_out_kernel<<<BSZ * TGT, 256, 0, stream>>>(logits, gen_ws, out_voc);
  copy_prob_kernel<<<(BSZ * TGT * AGENTS * SRC / 4) / 256, 256, 0, stream>>>(
      (const float4*)attn_f, gen_ws, (float4*)out_copy);
}
